// SpikeGPT_54391465836811
// MI455X (gfx1250) — matrix operand order repaired and checked by reading, not compiled, not run on hardware
//
#include <hip/hip_runtime.h>
#include <hip/hip_bf16.h>

#define Ld 12
#define Dd 1024
#define Fd 4096
#define Td 128
#define Vd 50257
#define NT 256
#define NTILES ((Vd + 15) / 16)   // 3142 vocab tiles of 16 rows

typedef __attribute__((ext_vector_type(16))) __bf16 v16bf;
typedef __attribute__((ext_vector_type(8)))  float  v8f;

// ---------- reduction helpers ----------
__device__ __forceinline__ float blk_reduce(float v, float* sred) {
    int tid = threadIdx.x;
    sred[tid] = v;
    __syncthreads();
    #pragma unroll
    for (int s = NT / 2; s > 0; s >>= 1) {
        if (tid < s) sred[tid] += sred[tid + s];
        __syncthreads();
    }
    float r = sred[0];
    __syncthreads();
    return r;
}

__device__ __forceinline__ float wave_reduce(float v) {
    #pragma unroll
    for (int off = 16; off > 0; off >>= 1) v += __shfl_xor(v, off, 32);
    return v;
}

// one wave computes dot(w_row, vec); vec lives in LDS, w streamed float4-coalesced
__device__ __forceinline__ float dot_row4(const float4* __restrict__ w,
                                          const float4* v, int n4, int lane) {
    float s = 0.f;
    for (int j = lane; j < n4; j += 32) {
        float4 a = w[j];
        float4 b = v[j];
        s = fmaf(a.x, b.x, s); s = fmaf(a.y, b.y, s);
        s = fmaf(a.z, b.z, s); s = fmaf(a.w, b.w, s);
    }
    return wave_reduce(s);
}

__device__ __forceinline__ float sigmoidf_(float x) { return 1.f / (1.f + __expf(-x)); }

// ---------- state init (ws is poisoned 0xAA; re-init every call) ----------
__global__ void k_init(float* zstart, int nz, float* ppstart, int np) {
    int i = blockIdx.x * blockDim.x + threadIdx.x;
    if (i < nz) zstart[i] = 0.f;
    if (i < np) ppstart[i] = -1e30f;
}

// ---------- K0: x = ln0(emb[tok] + pos_emb[t]) ----------
__global__ __launch_bounds__(NT) void k_token_start(
    const int* __restrict__ tokens, int t,
    const float* __restrict__ emb, const float* __restrict__ pos_emb,
    const float* __restrict__ w, const float* __restrict__ b,
    float* __restrict__ x) {
    __shared__ float sred[NT];
    int tid = threadIdx.x;
    int tok = tokens[t];
    float xv[Dd / NT]; float s = 0.f, sq = 0.f;
    #pragma unroll
    for (int i = 0; i < Dd / NT; i++) {
        int j = tid + NT * i;
        float val = emb[(size_t)tok * Dd + j] + pos_emb[(size_t)t * Dd + j];
        xv[i] = val; s += val; sq += val * val;
    }
    float mean = blk_reduce(s, sred) * (1.f / Dd);
    float var  = blk_reduce(sq, sred) * (1.f / Dd) - mean * mean;
    float rstd = rsqrtf(var + 1e-5f);
    #pragma unroll
    for (int i = 0; i < Dd / NT; i++) {
        int j = tid + NT * i;
        x[j] = (xv[i] - mean) * rstd * w[j] + b[j];
    }
}

// ---------- K1: ln1 + time-mix + GEMV k,v,r (3072 rows; 1 wave/row) ----------
__global__ __launch_bounds__(NT) void k_sa_qkv(
    const float* __restrict__ x, const float* __restrict__ sx,
    const float* __restrict__ lw, const float* __restrict__ lb,
    const float* __restrict__ tmk, const float* __restrict__ tmv, const float* __restrict__ tmr,
    const float* __restrict__ kw, const float* __restrict__ vw, const float* __restrict__ rw,
    float* __restrict__ kout, float* __restrict__ vout, float* __restrict__ rout,
    float* __restrict__ xa_buf) {
    __shared__ __align__(16) float s_xk[Dd];
    __shared__ __align__(16) float s_xv[Dd];
    __shared__ __align__(16) float s_xr[Dd];
    __shared__ float sred[NT];
    int tid = threadIdx.x;
    float xv_[Dd / NT]; float s = 0.f, sq = 0.f;
    #pragma unroll
    for (int i = 0; i < Dd / NT; i++) {
        int j = tid + NT * i; float v = x[j];
        xv_[i] = v; s += v; sq += v * v;
    }
    float mean = blk_reduce(s, sred) * (1.f / Dd);
    float var  = blk_reduce(sq, sred) * (1.f / Dd) - mean * mean;
    float rstd = rsqrtf(var + 1e-5f);
    #pragma unroll
    for (int i = 0; i < Dd / NT; i++) {
        int j = tid + NT * i;
        float xa = (xv_[i] - mean) * rstd * lw[j] + lb[j];
        float sv = sx[j];
        float mk = tmk[j], mv = tmv[j], mr = tmr[j];
        s_xk[j] = xa * mk + sv * (1.f - mk);
        s_xv[j] = xa * mv + sv * (1.f - mv);
        s_xr[j] = xa * mr + sv * (1.f - mr);
        if (blockIdx.x == 0) xa_buf[j] = xa;   // next-token sx_att, copied by K2
    }
    __syncthreads();
    int wave = tid >> 5, lane = tid & 31;
    int g = blockIdx.x * 8 + wave;             // 0..3071
    const float* wrow; const float4* vec; float* outp; bool sig = false;
    if (g < Dd)            { wrow = kw + (size_t)g * Dd;           vec = (const float4*)s_xk; outp = kout + g; }
    else if (g < 2 * Dd)   { int r = g - Dd;   wrow = vw + (size_t)r * Dd; vec = (const float4*)s_xv; outp = vout + r; }
    else                   { int r = g - 2*Dd; wrow = rw + (size_t)r * Dd; vec = (const float4*)s_xr; outp = rout + r; sig = true; }
    float sum = dot_row4((const float4*)wrow, vec, Dd / 4, lane);
    if (lane == 0) *outp = sig ? sigmoidf_(sum) : sum;
}

// ---------- K2: WKV elementwise (redundant/block) + state update (sliced) + ow GEMV + LIF ----------
__global__ __launch_bounds__(NT) void k_sa_wkv_out(
    float* __restrict__ x,
    const float* __restrict__ kk, const float* __restrict__ vv, const float* __restrict__ rr,
    const float* __restrict__ tf, const float* __restrict__ tdr,
    const float* __restrict__ ow,
    const float* __restrict__ aap, const float* __restrict__ bbp, const float* __restrict__ ppp,
    float* __restrict__ aan, float* __restrict__ bbn, float* __restrict__ ppn,
    float* __restrict__ v1, float* __restrict__ sx_att_l, const float* __restrict__ xa_buf) {
    __shared__ __align__(16) float s_rwkv[Dd];
    int tid = threadIdx.x;
    int base = blockIdx.x * 8;
    #pragma unroll
    for (int i = 0; i < Dd / NT; i++) {
        int j = tid + NT * i;
        float k = kk[j], v = vv[j];
        float aa = aap[j], bb = bbp[j], pp = ppp[j];
        float ww = tf[j] + k;
        float q  = fmaxf(pp, ww);
        float e1 = __expf(pp - q), e2 = __expf(ww - q);
        float a = e1 * aa + e2 * v;
        float b = e1 * bb + e2;
        s_rwkv[j] = rr[j] * (a / b);
        if ((unsigned)(j - base) < 8u) {       // ping-pong state: this block's slice only
            float td  = -__expf(tdr[j]);
            float ww2 = pp + td;
            float q2  = fmaxf(ww2, k);
            float f1 = __expf(ww2 - q2), f2 = __expf(k - q2);
            aan[j] = f1 * aa + f2 * v;
            bbn[j] = f1 * bb + f2;
            ppn[j] = q2;
            sx_att_l[j] = xa_buf[j];
        }
    }
    __syncthreads();
    int wave = tid >> 5, lane = tid & 31;
    int row = base + wave;
    float sum = dot_row4((const float4*)(ow + (size_t)row * Dd), (const float4*)s_rwkv, Dd / 4, lane);
    if (lane == 0) {                           // LIF(tau=2, vth=1, hard reset)
        float vm = v1[row];
        vm += (sum - vm) * 0.5f;
        float sp = (vm >= 1.f) ? 1.f : 0.f;
        v1[row] = vm * (1.f - sp);
        x[row]  = x[row] + sp;
    }
}

// ---------- K3: ln2 + channel-mix + GEMV fkw (F rows, relu^2) & frw (D rows, sigmoid) ----------
__global__ __launch_bounds__(NT) void k_ff_kr(
    const float* __restrict__ x, const float* __restrict__ sx,
    const float* __restrict__ lw, const float* __restrict__ lb,
    const float* __restrict__ tmk, const float* __restrict__ tmr,
    const float* __restrict__ fkw, const float* __restrict__ frw,
    float* __restrict__ fkout, float* __restrict__ frout, float* __restrict__ xf_buf) {
    __shared__ __align__(16) float s_xk[Dd];
    __shared__ __align__(16) float s_xr[Dd];
    __shared__ float sred[NT];
    int tid = threadIdx.x;
    float xv_[Dd / NT]; float s = 0.f, sq = 0.f;
    #pragma unroll
    for (int i = 0; i < Dd / NT; i++) {
        int j = tid + NT * i; float v = x[j];
        xv_[i] = v; s += v; sq += v * v;
    }
    float mean = blk_reduce(s, sred) * (1.f / Dd);
    float var  = blk_reduce(sq, sred) * (1.f / Dd) - mean * mean;
    float rstd = rsqrtf(var + 1e-5f);
    #pragma unroll
    for (int i = 0; i < Dd / NT; i++) {
        int j = tid + NT * i;
        float xf = (xv_[i] - mean) * rstd * lw[j] + lb[j];
        float sv = sx[j];
        float mk = tmk[j], mr = tmr[j];
        s_xk[j] = xf * mk + sv * (1.f - mk);
        s_xr[j] = xf * mr + sv * (1.f - mr);
        if (blockIdx.x == 0) xf_buf[j] = xf;   // next-token sx_ff, copied by K4
    }
    __syncthreads();
    int wave = tid >> 5, lane = tid & 31;
    int g = blockIdx.x * 8 + wave;             // 0..5119
    if (g < Fd) {
        float sum = dot_row4((const float4*)(fkw + (size_t)g * Dd), (const float4*)s_xk, Dd / 4, lane);
        if (lane == 0) { float rl = fmaxf(sum, 0.f); fkout[g] = rl * rl; }
    } else {
        int r = g - Fd;
        float sum = dot_row4((const float4*)(frw + (size_t)r * Dd), (const float4*)s_xr, Dd / 4, lane);
        if (lane == 0) frout[r] = sigmoidf_(sum);
    }
}

// ---------- K4: GEMV fvw (D rows over F) + LIF + rescale ----------
__global__ __launch_bounds__(NT) void k_ff_out(
    float* __restrict__ x,
    const float* __restrict__ fk, const float* __restrict__ fr,
    const float* __restrict__ fvw,
    float* __restrict__ v2, const float* __restrict__ xf_buf, float* __restrict__ sx_ff_l,
    int rescale) {
    __shared__ __align__(16) float s_fk[Fd];   // 16 KB stage, shared by 8 row-waves
    int tid = threadIdx.x;
    #pragma unroll
    for (int i = 0; i < Fd / NT; i++) s_fk[tid + NT * i] = fk[tid + NT * i];
    __syncthreads();
    int wave = tid >> 5, lane = tid & 31;
    int row = blockIdx.x * 8 + wave;
    float sum = dot_row4((const float4*)(fvw + (size_t)row * Fd), (const float4*)s_fk, Fd / 4, lane);
    if (lane == 0) {
        float y = fr[row] * sum;
        float vm = v2[row];
        vm += (y - vm) * 0.5f;
        float sp = (vm >= 1.f) ? 1.f : 0.f;
        v2[row] = vm * (1.f - sp);
        float xn = x[row] + sp;
        if (rescale) xn *= 0.5f;               // (layer+1)%6==0 -> x*0.5
        x[row] = xn;
        sx_ff_l[row] = xf_buf[row];
    }
}

// ---------- K5: X[t] = ln_out(x) ----------
__global__ __launch_bounds__(NT) void k_token_end(
    const float* __restrict__ x, const float* __restrict__ w, const float* __restrict__ b,
    float* __restrict__ Xrow) {
    __shared__ float sred[NT];
    int tid = threadIdx.x;
    float xv[Dd / NT]; float s = 0.f, sq = 0.f;
    #pragma unroll
    for (int i = 0; i < Dd / NT; i++) {
        int j = tid + NT * i; float v = x[j];
        xv[i] = v; s += v; sq += v * v;
    }
    float mean = blk_reduce(s, sred) * (1.f / Dd);
    float var  = blk_reduce(sq, sred) * (1.f / Dd) - mean * mean;
    float rstd = rsqrtf(var + 1e-5f);
    #pragma unroll
    for (int i = 0; i < Dd / NT; i++) {
        int j = tid + NT * i;
        Xrow[j] = (xv[i] - mean) * rstd * w[j] + b[j];
    }
}

// ---------- Head GEMM: logits[T,V] = X[T,D] @ head^T, bf16 WMMA, f32 accumulate ----------
// A-frag (16x32 bf16, ISA 7.12.2): lane m=l&15, hi=l>>4; VGPR v pair -> K = (v<4?0:16)+(v&3)*2+hi*8
//   => per lane: K in [k0+hi*8, +8) then [k0+16+hi*8, +8)
// B-frag (32x16): lane n=l&15; same K order as A: K in [k0+hi*8, +8) then [k0+16+hi*8, +8)
// C/D (16x16 f32): lane n=l&15; element r -> M = r + hi*8
__global__ __launch_bounds__(128) void k_head_gemm(
    const float* __restrict__ head, const float* __restrict__ X, float* __restrict__ out) {
    int tid = threadIdx.x;
    int lane = tid & 31;
    int wtile = blockIdx.x * 4 + (tid >> 5);
    if (wtile >= NTILES) return;               // whole wave exits together (EXEC stays full)
    int vbase = wtile * 16;
    int m  = lane & 15;
    int hi = lane >> 4;
    int arowi = vbase + m;
    bool rowok = arowi < Vd;
    const float* arow = head + (size_t)(rowok ? arowi : 0) * Dd;

    v8f c[8];
    #pragma unroll
    for (int nt = 0; nt < 8; nt++)
        #pragma unroll
        for (int r = 0; r < 8; r++) c[nt][r] = 0.f;

    for (int k0 = 0; k0 < Dd; k0 += 32) {
        float af[16];
        if (rowok) {
            const float4* a0 = (const float4*)(arow + k0 + hi * 8);
            const float4* a1 = (const float4*)(arow + k0 + 16 + hi * 8);
            float4 p0 = a0[0], p1 = a0[1], p2 = a1[0], p3 = a1[1];
            af[0]=p0.x; af[1]=p0.y; af[2]=p0.z;  af[3]=p0.w;
            af[4]=p1.x; af[5]=p1.y; af[6]=p1.z;  af[7]=p1.w;
            af[8]=p2.x; af[9]=p2.y; af[10]=p2.z; af[11]=p2.w;
            af[12]=p3.x; af[13]=p3.y; af[14]=p3.z; af[15]=p3.w;
        } else {
            #pragma unroll
            for (int i = 0; i < 16; i++) af[i] = 0.f;
        }
        v16bf a;
        #pragma unroll
        for (int i = 0; i < 16; i++) a[i] = (__bf16)af[i];

        #pragma unroll
        for (int nt = 0; nt < 8; nt++) {
            int tcol = nt * 16 + (lane & 15);
            const float4* b4 = (const float4*)(X + (size_t)tcol * Dd + k0 + hi * 8);
            const float4* b5 = (const float4*)(X + (size_t)tcol * Dd + k0 + 16 + hi * 8);
            float4 q0 = b4[0], q1 = b4[1], q2 = b5[0], q3 = b5[1];
            float bf[16] = { q0.x,q0.y,q0.z,q0.w, q1.x,q1.y,q1.z,q1.w,
                             q2.x,q2.y,q2.z,q2.w, q3.x,q3.y,q3.z,q3.w };
            v16bf b;
            #pragma unroll
            for (int i = 0; i < 16; i++) b[i] = (__bf16)bf[i];
            c[nt] = __builtin_amdgcn_wmma_f32_16x16x32_bf16(
                false, a, false, b, (short)0, c[nt], false, false);
        }
    }
    int n = lane & 15;
    #pragma unroll
    for (int nt = 0; nt < 8; nt++) {
        int tok = nt * 16 + n;
        #pragma unroll
        for (int r = 0; r < 8; r++) {
            int orow = vbase + r + hi * 8;
            if (orow < Vd) out[(size_t)tok * Vd + orow] = c[nt][r];
        }
    }
}

// ---------- host ----------
extern "C" void kernel_launch(void* const* d_in, const int* in_sizes, int n_in,
                              void* d_out, int out_size, void* d_ws, size_t ws_size,
                              hipStream_t stream) {
    (void)in_sizes; (void)n_in; (void)out_size; (void)ws_size;
    const int*   tokens   = (const int*)  d_in[0];
    const float* emb      = (const float*)d_in[1];
    const float* pos_emb  = (const float*)d_in[2];
    const float* ln0_w    = (const float*)d_in[3];
    const float* ln0_b    = (const float*)d_in[4];
    const float* ln1_w    = (const float*)d_in[5];
    const float* ln1_b    = (const float*)d_in[6];
    const float* ln2_w    = (const float*)d_in[7];
    const float* ln2_b    = (const float*)d_in[8];
    const float* att_tmk  = (const float*)d_in[9];
    const float* att_tmv  = (const float*)d_in[10];
    const float* att_tmr  = (const float*)d_in[11];
    const float* time_first = (const float*)d_in[12];
    const float* tdr      = (const float*)d_in[13];
    const float* att_kw   = (const float*)d_in[14];
    const float* att_vw   = (const float*)d_in[15];
    const float* att_rw   = (const float*)d_in[16];
    const float* att_ow   = (const float*)d_in[17];
    const float* ffn_tmk  = (const float*)d_in[18];
    const float* ffn_tmr  = (const float*)d_in[19];
    const float* ffn_kw   = (const float*)d_in[20];
    const float* ffn_vw   = (const float*)d_in[21];
    const float* ffn_rw   = (const float*)d_in[22];
    const float* lnout_w  = (const float*)d_in[23];
    const float* lnout_b  = (const float*)d_in[24];
    const float* headw    = (const float*)d_in[25];

    float* ws = (float*)d_ws;
    float* x      = ws;                 // D
    float* xa_buf = x + Dd;             // D
    float* xf_buf = xa_buf + Dd;        // D
    float* kbuf   = xf_buf + Dd;        // D
    float* vbuf   = kbuf + Dd;          // D
    float* rbuf   = vbuf + Dd;          // D
    float* frbuf  = rbuf + Dd;          // D
    float* fkbuf  = frbuf + Dd;         // F
    float* sx_att = fkbuf + Fd;         // L*D   (zero-init span begins here)
    float* sx_ff  = sx_att + Ld * Dd;   // L*D
    float* v1     = sx_ff + Ld * Dd;    // L*D
    float* v2     = v1 + Ld * Dd;       // L*D
    float* aa     = v2 + Ld * Dd;       // 2*L*D
    float* bb     = aa + 2 * Ld * Dd;   // 2*L*D  (zero-init span ends here)
    float* pp     = bb + 2 * Ld * Dd;   // 2*L*D  -> -1e30 init
    float* X      = pp + 2 * Ld * Dd;   // T*D

    int nz = 8 * Ld * Dd;               // sx_att..bb
    int np = 2 * Ld * Dd;
    k_init<<<(nz + NT - 1) / NT, NT, 0, stream>>>(sx_att, nz, pp, np);

    for (int t = 0; t < Td; t++) {
        k_token_start<<<1, NT, 0, stream>>>(tokens, t, emb, pos_emb, ln0_w, ln0_b, x);
        int rb = t & 1, wb = 1 - rb;    // ping-pong WKV state per token parity
        for (int l = 0; l < Ld; l++) {
            size_t oD  = (size_t)l * Dd;
            size_t oDD = (size_t)l * Dd * Dd;
            size_t oFD = (size_t)l * Fd * Dd;
            k_sa_qkv<<<3 * Dd / 8, NT, 0, stream>>>(
                x, sx_att + oD, ln1_w + oD, ln1_b + oD,
                att_tmk + oD, att_tmv + oD, att_tmr + oD,
                att_kw + oDD, att_vw + oDD, att_rw + oDD,
                kbuf, vbuf, rbuf, xa_buf);
            k_sa_wkv_out<<<Dd / 8, NT, 0, stream>>>(
                x, kbuf, vbuf, rbuf, time_first + oD, tdr + oD, att_ow + oDD,
                aa + (size_t)rb * Ld * Dd + oD, bb + (size_t)rb * Ld * Dd + oD, pp + (size_t)rb * Ld * Dd + oD,
                aa + (size_t)wb * Ld * Dd + oD, bb + (size_t)wb * Ld * Dd + oD, pp + (size_t)wb * Ld * Dd + oD,
                v1 + oD, sx_att + oD, xa_buf);
            k_ff_kr<<<(Fd + Dd) / 8, NT, 0, stream>>>(
                x, sx_ff + oD, ln2_w + oD, ln2_b + oD,
                ffn_tmk + oD, ffn_tmr + oD,
                ffn_kw + oFD, ffn_rw + oDD,
                fkbuf, frbuf, xf_buf);
            k_ff_out<<<Dd / 8, NT, 0, stream>>>(
                x, fkbuf, frbuf, ffn_vw + oFD,
                v2 + oD, xf_buf, sx_ff + oD, ((l + 1) % 6 == 0) ? 1 : 0);
        }
        k_token_end<<<1, NT, 0, stream>>>(x, lnout_w, lnout_b, X + (size_t)t * Dd);
    }

    // batched head projection over all 128 tokens: the only dense GEMM -> WMMA bf16
    k_head_gemm<<<(NTILES + 3) / 4, 128, 0, stream>>>(headw, X, (float*)d_out);
}
